// SRULayer_84748294685271
// MI455X (gfx1250) — compile-verified, matrix operand order checked by reading, not run on hardware
//
#include <hip/hip_runtime.h>
#include <math.h>

// ---------------------------------------------------------------------------
// Types for WMMA (CDNA5 / gfx1250, wave32)
// ---------------------------------------------------------------------------
typedef __attribute__((ext_vector_type(16))) _Float16 v16h;
typedef __attribute__((ext_vector_type(8)))  _Float16 v8h;
typedef __attribute__((ext_vector_type(4)))  _Float16 v4h;
typedef __attribute__((ext_vector_type(8)))  float    v8f;

// Problem constants
#define LEN   2048
#define BATCH 8
#define DIM   1024
#define N3    3072          // 3*DIM
#define ROWS  16384         // LEN*BATCH
#define CH    8192          // BATCH*DIM
#define NSEG  16
#define SEGT  128           // LEN / NSEG

// Fast transcendentals: v_exp_f32 + v_rcp_f32 instead of IEEE division chain
__device__ __forceinline__ float fast_sigmoid(float v) {
    return __builtin_amdgcn_rcpf(1.0f + __expf(-v));
}
__device__ __forceinline__ float fast_tanh(float c) {
    // tanh(c) = 2*sigmoid(2c) - 1
    return 2.0f * fast_sigmoid(2.0f * c) - 1.0f;
}

// ---------------------------------------------------------------------------
// Kernel 1: W (D x 3D fp32, row-major) -> Wt (3D x D f16, row-major = N-major)
// ---------------------------------------------------------------------------
__global__ void wt_kernel(const float* __restrict__ W, _Float16* __restrict__ Wt) {
    __shared__ float tile[32][33];
    int n0 = blockIdx.x * 32;
    int k0 = blockIdx.y * 32;
    int tx = threadIdx.x;   // 0..31
    int ty = threadIdx.y;   // 0..7
#pragma unroll
    for (int i = 0; i < 4; i++)
        tile[ty + i * 8][tx] = W[(size_t)(k0 + ty + i * 8) * N3 + n0 + tx];
    __syncthreads();
#pragma unroll
    for (int i = 0; i < 4; i++)
        Wt[(size_t)(n0 + ty + i * 8) * DIM + k0 + tx] = (_Float16)tile[tx][ty + i * 8];
}

// ---------------------------------------------------------------------------
// Kernel 2: LayerNorm over D, write x_norm as f16 (GEMM A operand + blend src)
// ---------------------------------------------------------------------------
__global__ void ln_kernel(const float* __restrict__ x,
                          const float* __restrict__ gamma,
                          const float* __restrict__ beta,
                          _Float16* __restrict__ xn) {
    int row = blockIdx.x;                 // 0..16383
    int t   = threadIdx.x;                // 0..255
    const float* xr = x + (size_t)row * DIM;
    float4 v = *(const float4*)(xr + t * 4);
    float s  = v.x + v.y + v.z + v.w;
    float sq = v.x * v.x + v.y * v.y + v.z * v.z + v.w * v.w;

    __shared__ float rs[256];
    __shared__ float rq[256];
    rs[t] = s; rq[t] = sq;
    __syncthreads();
    for (int off = 128; off > 0; off >>= 1) {
        if (t < off) { rs[t] += rs[t + off]; rq[t] += rq[t + off]; }
        __syncthreads();
    }
    float mu  = rs[0] * (1.0f / DIM);
    float var = rq[0] * (1.0f / DIM) - mu * mu;
    float inv = rsqrtf(var + 1e-5f);

    float4 g  = *(const float4*)(gamma + t * 4);
    float4 be = *(const float4*)(beta  + t * 4);
    v4h o;
    o[0] = (_Float16)((v.x - mu) * inv * g.x + be.x);
    o[1] = (_Float16)((v.y - mu) * inv * g.y + be.y);
    o[2] = (_Float16)((v.z - mu) * inv * g.z + be.z);
    o[3] = (_Float16)((v.w - mu) * inv * g.w + be.w);
    *(v4h*)(xn + (size_t)row * DIM + t * 4) = o;
}

// ---------------------------------------------------------------------------
// Kernel 3: WMMA GEMM  C[16384 x 3072] = xn(f16) @ Wt^T(f16) + b, fused gates.
// Block tile 128x256, BK=32, 8 waves (2 M x 4 N), wave tile 64x64,
// 4x4 v_wmma_f32_16x16x32_f16 accumulators per wave (16 WMMA per K step,
// 1:1 WMMA : ds_load_b128 ratio).
// ---------------------------------------------------------------------------
// Build a 16-bit A/B fragment per the CDNA5 ISA 16x32 layout:
//   lanes 0-15 : K = {0..7, 16..23};  lanes 16-31 : K = {8..15, 24..31}
__device__ __forceinline__ v16h ld_frag(const _Float16* rowp, int hi) {
    v8h lo  = *(const v8h*)(rowp + hi * 8);
    v8h hi8 = *(const v8h*)(rowp + 16 + hi * 8);
    return __builtin_shufflevector(lo, hi8, 0, 1, 2, 3, 4, 5, 6, 7,
                                            8, 9, 10, 11, 12, 13, 14, 15);
}

__global__ void __launch_bounds__(256)
gemm_kernel(const _Float16* __restrict__ xn, const _Float16* __restrict__ Wt,
            const float* __restrict__ bias,
            _Float16* __restrict__ ub, _Float16* __restrict__ fgb,
            _Float16* __restrict__ rgb) {
    __shared__ __align__(16) _Float16 As[128][40];   // +8 f16 pad (80B rows)
    __shared__ __align__(16) _Float16 Bs[256][40];

    int t      = threadIdx.x;
    int blockN = blockIdx.x;      // 0..11
    int blockM = blockIdx.y;      // 0..127
    int wave   = t >> 5;
    int lane   = t & 31;
    int waveM  = wave >> 2;       // 0..1 -> M offset *64
    int waveN  = wave & 3;        // 0..3 -> N offset *64
    int lrow   = t >> 1;          // 0..127 : A row loaded by this thread
    int lchunk = t & 1;           // 0..1   : 16-f16 chunk of A row

    const _Float16* aG = xn + (size_t)(blockM * 128 + lrow) * DIM + lchunk * 16;
    const _Float16* bG = Wt + (size_t)(blockN * 256 + t) * DIM;   // 1 B row/thread

    v8f acc[4][4] = {};
    int mrow = lane & 15;
    int hi   = lane >> 4;

    for (int k0 = 0; k0 < DIM; k0 += 32) {
        float4 a0 = *(const float4*)(aG + k0);
        float4 a1 = *(const float4*)(aG + k0 + 8);
        float4 b0 = *(const float4*)(bG + k0);
        float4 b1 = *(const float4*)(bG + k0 + 8);
        float4 b2 = *(const float4*)(bG + k0 + 16);
        float4 b3 = *(const float4*)(bG + k0 + 24);
        // hint next K tile into cache (global_prefetch_b8 on gfx1250)
        __builtin_prefetch(aG + k0 + 32, 0, 3);
        __builtin_prefetch(bG + k0 + 32, 0, 3);
        __syncthreads();                     // previous iter done reading LDS
        *(float4*)&As[lrow][lchunk * 16]     = a0;
        *(float4*)&As[lrow][lchunk * 16 + 8] = a1;
        *(float4*)&Bs[t][0]  = b0;
        *(float4*)&Bs[t][8]  = b1;
        *(float4*)&Bs[t][16] = b2;
        *(float4*)&Bs[t][24] = b3;
        __syncthreads();

        v16h bf[4];
#pragma unroll
        for (int nf = 0; nf < 4; nf++)
            bf[nf] = ld_frag(&Bs[waveN * 64 + nf * 16 + mrow][0], hi);
#pragma unroll
        for (int mi = 0; mi < 4; mi++) {
            v16h af = ld_frag(&As[waveM * 64 + mi * 16 + mrow][0], hi);
#pragma unroll
            for (int nf = 0; nf < 4; nf++)
                acc[mi][nf] = __builtin_amdgcn_wmma_f32_16x16x32_f16(
                    false, af, false, bf[nf], (short)0, acc[mi][nf], false, false);
        }
    }

    // Epilogue: bias + gate nonlinearity, split N into u / f / r regions.
    // C/D layout: lane n = lane&15, VGPR j -> m = j + (lane>>4)*8
    int nloc = lane & 15;
    int mhi  = lane >> 4;
#pragma unroll
    for (int mi = 0; mi < 4; mi++) {
        int gmb = blockM * 128 + waveM * 64 + mi * 16 + mhi * 8;
#pragma unroll
        for (int nf = 0; nf < 4; nf++) {
            int gn     = blockN * 256 + waveN * 64 + nf * 16 + nloc;
            float bv   = bias[gn];
            int region = gn >> 10;      // uniform per 16-wide frag (tiles 256-aligned)
            int ncol   = gn & 1023;
            _Float16* dst = (region == 0) ? ub : ((region == 1) ? fgb : rgb);
#pragma unroll
            for (int j = 0; j < 8; j++) {
                float v = acc[mi][nf][j] + bv;
                if (region) v = fast_sigmoid(v);     // sigmoid for f,r
                dst[(size_t)(gmb + j) * DIM + ncol] = (_Float16)v;
            }
        }
    }
}

// ---------------------------------------------------------------------------
// Kernels 4-6: segmented linear recurrence  c_t = f*c_{t-1} + (1-f)*u
// ---------------------------------------------------------------------------
// Pass 1: per (segment, channel): P = prod f, Q = response from zero state
__global__ void seg_reduce_kernel(const _Float16* __restrict__ fg,
                                  const _Float16* __restrict__ ub,
                                  float* __restrict__ P, float* __restrict__ Q) {
    int tid = blockIdx.x * blockDim.x + threadIdx.x;  // 0 .. 16*8192-1
    int s = tid >> 13;
    int j = tid & (CH - 1);
    float p = 1.0f, q = 0.0f;
    for (int tt = 0; tt < SEGT; tt++) {
        int idx = (s * SEGT + tt) * CH + j;
        float f = (float)fg[idx];
        float u = (float)ub[idx];
        p *= f;
        q = f * q + (1.0f - f) * u;
    }
    P[tid] = p;
    Q[tid] = q;
}

// Pass 2: tiny sequential prefix over 16 segments per channel; emits last_c
__global__ void seg_scan_kernel(const float* __restrict__ c0,
                                const float* __restrict__ P,
                                const float* __restrict__ Q,
                                float* __restrict__ cstart,
                                float* __restrict__ last_c) {
    int j = blockIdx.x * blockDim.x + threadIdx.x;    // 0..8191
    float c = c0[j];
#pragma unroll
    for (int s = 0; s < NSEG; s++) {
        cstart[s * CH + j] = c;
        c = P[s * CH + j] * c + Q[s * CH + j];
    }
    last_c[j] = c;
}

// Pass 3: replay each segment from its true start; blend + residual
__global__ void seg_apply_kernel(const float* __restrict__ x,
                                 const _Float16* __restrict__ ub,
                                 const _Float16* __restrict__ fg,
                                 const _Float16* __restrict__ rg,
                                 const _Float16* __restrict__ xn,
                                 const float* __restrict__ cstart,
                                 float* __restrict__ out) {
    int tid = blockIdx.x * blockDim.x + threadIdx.x;
    int s = tid >> 13;
    int j = tid & (CH - 1);
    float c = cstart[s * CH + j];
    for (int tt = 0; tt < SEGT; tt++) {
        int idx = (s * SEGT + tt) * CH + j;
        float f  = (float)fg[idx];
        float u  = (float)ub[idx];
        float r  = (float)rg[idx];
        float xv = (float)xn[idx];
        c = f * c + (1.0f - f) * u;
        float hs = r * fast_tanh(c) + (1.0f - r) * xv;
        out[idx] = x[idx] + hs;
    }
}

// ---------------------------------------------------------------------------
// Launcher
// ---------------------------------------------------------------------------
extern "C" void kernel_launch(void* const* d_in, const int* in_sizes, int n_in,
                              void* d_out, int out_size, void* d_ws, size_t ws_size,
                              hipStream_t stream) {
    const float* x     = (const float*)d_in[0];   // (L,B,D)
    const float* c0    = (const float*)d_in[1];   // (B,D)
    const float* W     = (const float*)d_in[2];   // (D,3D)
    const float* b     = (const float*)d_in[3];   // (3D)
    const float* gamma = (const float*)d_in[4];   // (D)
    const float* beta  = (const float*)d_in[5];   // (D)

    char* ws = (char*)d_ws;
    // all offsets 16B-aligned
    _Float16* Wt  = (_Float16*)(ws + 0);              //  6 MiB : 3072x1024 f16
    _Float16* xn  = (_Float16*)(ws + 6291456);        // 32 MiB : x_norm f16
    _Float16* ub  = (_Float16*)(ws + 39845888);       // 32 MiB : u f16
    _Float16* fgb = (_Float16*)(ws + 73400320);       // 32 MiB : sigmoid(f) f16
    _Float16* rgb = (_Float16*)(ws + 106954752);      // 32 MiB : sigmoid(r) f16
    float* P      = (float*)(ws + 140509184);         // 512 KiB
    float* Q      = (float*)(ws + 141033472);         // 512 KiB
    float* cstart = (float*)(ws + 141557760);         // 512 KiB
    (void)ws_size; (void)in_sizes; (void)n_in; (void)out_size;

    float* out    = (float*)d_out;                    // (L,B,D)
    float* last_c = (float*)d_out + (size_t)ROWS * DIM;  // (B,D)

    // 1) W -> Wt (f16, N-major)
    wt_kernel<<<dim3(N3 / 32, DIM / 32), dim3(32, 8), 0, stream>>>(W, Wt);
    // 2) LayerNorm -> xn f16
    ln_kernel<<<ROWS, 256, 0, stream>>>(x, gamma, beta, xn);
    // 3) WMMA GEMM + fused bias/sigmoid epilogue
    gemm_kernel<<<dim3(N3 / 256, ROWS / 128), 256, 0, stream>>>(xn, Wt, b, ub, fgb, rgb);
    // 4-6) segmented scan + blend + residual
    seg_reduce_kernel<<<(NSEG * CH) / 256, 256, 0, stream>>>(fgb, ub, P, Q);
    seg_scan_kernel<<<CH / 256, 256, 0, stream>>>(c0, P, Q, cstart, last_c);
    seg_apply_kernel<<<(NSEG * CH) / 256, 256, 0, stream>>>(x, ub, fgb, rgb, xn, cstart, out);
}
